// FMoELinear_1958505087361
// MI455X (gfx1250) — matrix operand order repaired and checked by reading, not compiled, not run on hardware
//
#include <hip/hip_runtime.h>
#include <hip/hip_bf16.h>

typedef __bf16 bf16_t;
typedef __attribute__((ext_vector_type(16))) __bf16 v16bf;
typedef __attribute__((ext_vector_type(8)))  __bf16 v8bf;
typedef __attribute__((ext_vector_type(4)))  __bf16 v4bf;
typedef __attribute__((ext_vector_type(8)))  float  v8f;

#define NUM_EXPERT 32
#define IN_FEAT    256
#define OUT_FEAT   256
#define TILE_M     128
#define MSUB       (TILE_M / 16)      // 8 M-subtiles per wave
#define LDS_STRIDE 264   // bf16 elems per row: 256 + 8 pad -> 528B stride, bank-rotating
#define WS_TABLE_BYTES 4096

// ---------------------------------------------------------------------------
// Setup: build tile table in workspace.
// table[0..32]   : tilePrefix  (tiles before expert e; [32] = total tiles)
// table[33..65]  : startTok    (token start of expert e; [65] = total tokens)
// ---------------------------------------------------------------------------
__global__ void moe_setup_kernel(const long long* __restrict__ counts,
                                 int* __restrict__ table) {
    if (threadIdx.x == 0) {
        int start = 0, tp = 0;
        for (int e = 0; e < NUM_EXPERT; ++e) {
            int c = (int)counts[e];
            table[e]      = tp;
            table[33 + e] = start;
            tp    += (c + TILE_M - 1) / TILE_M;
            start += c;
        }
        table[32]      = tp;
        table[33 + 32] = start;
    }
}

// ---------------------------------------------------------------------------
// One-shot weight split: fp32 W -> bf16 hi/lo planes in workspace.
// ---------------------------------------------------------------------------
__global__ void moe_wsplit_kernel(const float* __restrict__ w,
                                  bf16_t* __restrict__ whi,
                                  bf16_t* __restrict__ wlo) {
    const size_t idx = ((size_t)blockIdx.x * 256 + threadIdx.x) * 4;
    const float4 v = *(const float4*)(w + idx);
    const float fv[4] = {v.x, v.y, v.z, v.w};
    v4bf h, l;
    #pragma unroll
    for (int j = 0; j < 4; ++j) {
        bf16_t hj = (bf16_t)fv[j];
        h[j] = hj;
        l[j] = (bf16_t)(fv[j] - (float)hj);
    }
    *(v4bf*)(whi + idx) = h;
    *(v4bf*)(wlo + idx) = l;
}

// ---------------------------------------------------------------------------
// Main grouped-GEMM kernel. Block = 256 threads = 8 waves.
// Block tile: M=128 tokens x N=256 (full OUT). Wave owns 32 output columns.
// fp32 emulated via split-bf16 (hi+lo) with 3 bf16 WMMAs per 16x16x32 step.
// PRE=true: B read from pre-split bf16 planes (no conversion VALU in loop).
// ---------------------------------------------------------------------------
template <bool PRE>
__launch_bounds__(256)
__global__ void moe_gemm_kernel(const float*  __restrict__ inp,
                                const float*  __restrict__ w_f32,
                                const bf16_t* __restrict__ whi,
                                const bf16_t* __restrict__ wlo,
                                const int*    __restrict__ table,
                                float*        __restrict__ out) {
    __shared__ bf16_t lds_hi[TILE_M * LDS_STRIDE];
    __shared__ bf16_t lds_lo[TILE_M * LDS_STRIDE];

    const int tile = blockIdx.x;
    const int totalTiles = table[32];
    if (tile >= totalTiles) return;

    // Uniform scan: which expert owns this tile (scalar loads, SALU)
    int e = 0;
    #pragma unroll
    for (int i = 1; i < NUM_EXPERT; ++i) e += (tile >= table[i]) ? 1 : 0;

    const int lt        = tile - table[e];
    const int segStart  = table[33 + e];
    const int segCount  = table[33 + e + 1] - segStart;
    const int m0        = segStart + lt * TILE_M;          // first token of tile
    const int rowsValid = min(TILE_M, segCount - lt * TILE_M);

    const int tid = threadIdx.x;

    // ---- Stage A tile (128x256 fp32) into LDS as split bf16 hi/lo planes ----
    #pragma unroll
    for (int it = 0; it < (TILE_M * IN_FEAT) / (256 * 4); ++it) {   // 32 iters
        const int idx = (it * 256 + tid) * 4;      // float index in tile
        const int r   = idx >> 8;                  // row 0..127
        const int c   = idx & (IN_FEAT - 1);       // col, multiple of 4
        float4 v = make_float4(0.f, 0.f, 0.f, 0.f);
        if (r < rowsValid)
            v = *(const float4*)(inp + (size_t)(m0 + r) * IN_FEAT + c);
        const float fv[4] = {v.x, v.y, v.z, v.w};
        v4bf hv, lv;
        #pragma unroll
        for (int j = 0; j < 4; ++j) {
            bf16_t h = (bf16_t)fv[j];
            hv[j] = h;
            lv[j] = (bf16_t)(fv[j] - (float)h);
        }
        const int o = r * LDS_STRIDE + c;          // 8B-aligned
        *(v4bf*)(&lds_hi[o]) = hv;
        *(v4bf*)(&lds_lo[o]) = lv;
    }
    __syncthreads();

    // ---- Per-wave compute ----
    const int wave = tid >> 5;
    const int lane = tid & 31;
    const int half = lane >> 4;      // 0: lanes 0-15, 1: lanes 16-31
    const int l16  = lane & 15;
    const int n0   = wave * 32;      // this wave's output-column base

    v8f acc[MSUB][2];
    #pragma unroll
    for (int ms = 0; ms < MSUB; ++ms)
        #pragma unroll
        for (int ns = 0; ns < 2; ++ns) {
            v8f z = {0.f, 0.f, 0.f, 0.f, 0.f, 0.f, 0.f, 0.f};
            acc[ms][ns] = z;
        }

    const size_t  ebase  = (size_t)e * OUT_FEAT * IN_FEAT;
    const float*  wf     = w_f32 + ebase;
    const bf16_t* bhbase = whi + ebase;
    const bf16_t* blbase = wlo + ebase;

    for (int kk = 0; kk < IN_FEAT; kk += 32) {
        // B fragments: B[k][n] = weight[e][n][k]; lane holds 16 contiguous k
        v16bf Bhi[2], Blo[2];
        #pragma unroll
        for (int ns = 0; ns < 2; ++ns) {
            const int n = n0 + ns * 16 + l16;
            if (PRE) {
                const size_t off = (size_t)n * IN_FEAT + kk + half * 16;
                const v8bf h0 = *(const v8bf*)(bhbase + off);
                const v8bf h1 = *(const v8bf*)(bhbase + off + 8);
                const v8bf l0 = *(const v8bf*)(blbase + off);
                const v8bf l1 = *(const v8bf*)(blbase + off + 8);
                Bhi[ns] = __builtin_shufflevector(h0, h1, 0,1,2,3,4,5,6,7,
                                                  8,9,10,11,12,13,14,15);
                Blo[ns] = __builtin_shufflevector(l0, l1, 0,1,2,3,4,5,6,7,
                                                  8,9,10,11,12,13,14,15);
                // Warm L0 for next K-step (weights are L2-resident);
                // emits global_prefetch_b8 (gfx1250 path).
                const size_t poff = (size_t)n * IN_FEAT +
                                    min(kk + 32, IN_FEAT - 32) + half * 16;
                __builtin_prefetch(bhbase + poff, 0, 3);
                __builtin_prefetch(blbase + poff, 0, 3);
            } else {
                const float* bp = wf + (size_t)n * IN_FEAT + kk + half * 16;
                const float4 f0 = ((const float4*)bp)[0];
                const float4 f1 = ((const float4*)bp)[1];
                const float4 f2 = ((const float4*)bp)[2];
                const float4 f3 = ((const float4*)bp)[3];
                const float fv[16] = {f0.x,f0.y,f0.z,f0.w, f1.x,f1.y,f1.z,f1.w,
                                      f2.x,f2.y,f2.z,f2.w, f3.x,f3.y,f3.z,f3.w};
                #pragma unroll
                for (int j = 0; j < 16; ++j) {
                    bf16_t h = (bf16_t)fv[j];
                    Bhi[ns][j] = h;
                    Blo[ns][j] = (bf16_t)(fv[j] - (float)h);
                }
            }
        }

        #pragma unroll
        for (int ms = 0; ms < MSUB; ++ms) {
            // A fragment: lane layout = k in [kk+half*16 .. +15]
            const int o = (ms * 16 + l16) * LDS_STRIDE + kk + half * 16;
            const v8bf a0 = *(const v8bf*)(&lds_hi[o]);
            const v8bf a1 = *(const v8bf*)(&lds_hi[o + 8]);
            const v8bf b0 = *(const v8bf*)(&lds_lo[o]);
            const v8bf b1 = *(const v8bf*)(&lds_lo[o + 8]);
            const v16bf Ahi = __builtin_shufflevector(a0, a1, 0,1,2,3,4,5,6,7,
                                                      8,9,10,11,12,13,14,15);
            const v16bf Alo = __builtin_shufflevector(b0, b1, 0,1,2,3,4,5,6,7,
                                                      8,9,10,11,12,13,14,15);
            #pragma unroll
            for (int ns = 0; ns < 2; ++ns) {
                acc[ms][ns] = __builtin_amdgcn_wmma_f32_16x16x32_bf16(
                    false, Ahi, false, Bhi[ns], (short)0, acc[ms][ns], false, false);
                acc[ms][ns] = __builtin_amdgcn_wmma_f32_16x16x32_bf16(
                    false, Alo, false, Bhi[ns], (short)0, acc[ms][ns], false, false);
                acc[ms][ns] = __builtin_amdgcn_wmma_f32_16x16x32_bf16(
                    false, Ahi, false, Blo[ns], (short)0, acc[ms][ns], false, false);
            }
        }
    }

    // ---- Store: C/D layout row = v + 8*half, col = lane%16 ----
    if (rowsValid == TILE_M) {
        // Fast path (vast majority of tiles): no per-row guards.
        #pragma unroll
        for (int ms = 0; ms < MSUB; ++ms) {
            float* op = out + (size_t)(m0 + ms * 16 + half * 8) * OUT_FEAT + n0 + l16;
            #pragma unroll
            for (int v = 0; v < 8; ++v) {
                op[0]  = acc[ms][0][v];
                op[16] = acc[ms][1][v];
                op += OUT_FEAT;
            }
        }
    } else {
        #pragma unroll
        for (int ms = 0; ms < MSUB; ++ms) {
            #pragma unroll
            for (int v = 0; v < 8; ++v) {
                const int r = ms * 16 + half * 8 + v;
                if (r < rowsValid) {
                    float* op = out + (size_t)(m0 + r) * OUT_FEAT + n0 + l16;
                    op[0]  = acc[ms][0][v];
                    op[16] = acc[ms][1][v];
                }
            }
        }
    }
}

// ---------------------------------------------------------------------------
extern "C" void kernel_launch(void* const* d_in, const int* in_sizes, int n_in,
                              void* d_out, int out_size, void* d_ws, size_t ws_size,
                              hipStream_t stream) {
    const float*     inp    = (const float*)d_in[0];
    const float*     weight = (const float*)d_in[1];
    const long long* counts = (const long long*)d_in[2];   // int64 per reference
    float* out = (float*)d_out;
    int*   table = (int*)d_ws;

    const int T = in_sizes[0] / IN_FEAT;
    const int maxTiles = T / TILE_M + NUM_EXPERT;   // padded upper bound

    const size_t wElems = (size_t)NUM_EXPERT * OUT_FEAT * IN_FEAT;
    bf16_t* whi = (bf16_t*)((char*)d_ws + WS_TABLE_BYTES);
    bf16_t* wlo = whi + wElems;
    const bool pre = ws_size >= WS_TABLE_BYTES + 2 * wElems * sizeof(bf16_t);

    moe_setup_kernel<<<1, 32, 0, stream>>>(counts, table);
    if (pre) {
        moe_wsplit_kernel<<<(int)(wElems / (256 * 4)), 256, 0, stream>>>(weight, whi, wlo);
        moe_gemm_kernel<true><<<maxTiles, 256, 0, stream>>>(inp, weight, whi, wlo, table, out);
    } else {
        moe_gemm_kernel<false><<<maxTiles, 256, 0, stream>>>(inp, weight, whi, wlo, table, out);
    }
}
